// Head_2937757631130
// MI455X (gfx1250) — matrix operand order repaired and checked by reading, not compiled, not run on hardware
//
#include <hip/hip_runtime.h>
#include <hip/hip_bf16.h>
#include <cstdint>
#include <cstddef>

// Problem constants (match reference: B=4, T=2048, D=1024, H=64)
constexpr int kB = 4;
constexpr int kT = 2048;
constexpr int kD = 1024;
constexpr int kH = 64;

typedef __attribute__((ext_vector_type(16))) __bf16 v16bf;
typedef __attribute__((ext_vector_type(8)))  float  v8f;
typedef int v4i_vs __attribute__((vector_size(16)));   // matches builtin param

union BF16Frag {
  v16bf    v;
  uint32_t u[8];
  __bf16   h[16];
};

// A-fragment (16-bit, 16x32) K index for dword j, half-wave hf:
//   contiguous: K = 16*hf + 2j, 2j+1
__device__ __forceinline__ int a_kbase(int j, int hf) {
  return (j << 1) + (hf << 4);
}

// ---------------------------------------------------------------------------
// Async global -> LDS 16-byte copy (gfx1250 GLOBAL_LOAD_ASYNC_TO_LDS_B128,
// tracked by ASYNCcnt).  Builtin when available, inline asm otherwise.
// ---------------------------------------------------------------------------
#if defined(__has_builtin) && __has_builtin(__builtin_amdgcn_global_load_async_to_lds_b128)
__device__ __forceinline__ void async_cp16(void* lds, const void* g) {
  __builtin_amdgcn_global_load_async_to_lds_b128(
      (__attribute__((address_space(1))) v4i_vs*)(uintptr_t)g,
      (__attribute__((address_space(3))) v4i_vs*)lds, 0, 0);
}
#else
__device__ __forceinline__ void async_cp16(void* lds, const void* g) {
  uint32_t loff = (uint32_t)(uintptr_t)(__attribute__((address_space(3))) void*)lds;
  uint64_t gaddr = (uint64_t)(uintptr_t)g;
  asm volatile("global_load_async_to_lds_b128 %0, %1, off"
               :: "v"(loff), "v"(gaddr) : "memory");
}
#endif

__device__ __forceinline__ void wait_async0() {
  asm volatile("s_wait_asynccnt 0x0" ::: "memory");
}

// ---------------------------------------------------------------------------
// Kernel 0: W [D,H] f32 (x3)  ->  Wt [3][H][D] bf16 (transposed, contiguous K)
// ---------------------------------------------------------------------------
__global__ void wt_kernel(const float* __restrict__ Wq,
                          const float* __restrict__ Wk,
                          const float* __restrict__ Wv,
                          __bf16* __restrict__ Wt) {
  int i = blockIdx.x * blockDim.x + threadIdx.x;
  if (i >= 3 * kD * kH) return;
  int m = i / (kD * kH);
  int r = i % (kD * kH);
  int h = r / kD;
  int d = r % kD;
  const float* W = (m == 0) ? Wq : ((m == 1) ? Wk : Wv);
  Wt[i] = (__bf16)W[d * kH + h];
}

// ---------------------------------------------------------------------------
// Kernel 1: QKV projection.  8 waves/block, one 16-row token tile per wave.
// The Wt k-slice (3 mats x 64 h x 32 k = 12 KB) is async-staged in LDS once
// per k-step and shared by all 8 waves (double-buffered).  B-fragments are
// software-pipelined (2-deep) so ds_loads overlap WMMA execution.
//   Q,K stored [B,T,H] bf16 ; V stored transposed [B,H,T] bf16.
// ---------------------------------------------------------------------------
__global__ __launch_bounds__(256) void qkv_kernel(
    const float* __restrict__ x, const __bf16* __restrict__ Wt,
    __bf16* __restrict__ Q, __bf16* __restrict__ K, __bf16* __restrict__ Vt) {
  __shared__ __bf16 Wl[2][3 * 64 * 32];   // 2 x 12 KB

  const int tid  = threadIdx.x;
  const int lane = tid & 31;
  const int wave = tid >> 5;
  const int tile = blockIdx.x * 8 + wave;   // 16-row tile over B*T rows
  const int row0 = tile * 16;
  const int hf   = lane >> 4;
  const int ln   = lane & 15;

  // stage Wt[.][.][k .. k+31] -> Wl[buf] : 768 x 16B chunks, 3 per thread
  auto stage_w = [&](int buf, int k) {
    #pragma unroll
    for (int r = 0; r < 3; ++r) {
      int idx = tid + 256 * r;          // 0..767
      int mh = idx >> 2, piece = idx & 3;
      async_cp16(&Wl[buf][mh * 32 + piece * 8],
                 Wt + (size_t)mh * kD + k + piece * 8);
    }
  };

  v8f acc[12];                           // [m*4 + t]
  #pragma unroll
  for (int i = 0; i < 12; ++i)
    acc[i] = v8f{0.f, 0.f, 0.f, 0.f, 0.f, 0.f, 0.f, 0.f};

  stage_w(0, 0);
  wait_async0();
  __syncthreads();

  int buf = 0;
  for (int k = 0; k < kD; k += 32) {
    if (k + 32 < kD) stage_w(buf ^ 1, k + 32);   // prefetch next slice

    // A fragment: 16x32 tile of x.  Issue all 4 b128 loads first (one
    // clause), then convert f32 -> bf16 with v_cvt_pk_bf16_f32.
    const float* xrow = x + (size_t)(row0 + ln) * kD + k + hf * 16;
    __builtin_prefetch(xrow + 32, 0, 1);
    float4 f0 = *(const float4*)(xrow + 0);
    float4 f1 = *(const float4*)(xrow + 4);
    float4 f2 = *(const float4*)(xrow + 8);
    float4 f3 = *(const float4*)(xrow + 12);
    BF16Frag a;
    a.h[0]  = (__bf16)f0.x; a.h[1]  = (__bf16)f0.y;
    a.h[2]  = (__bf16)f0.z; a.h[3]  = (__bf16)f0.w;
    a.h[4]  = (__bf16)f1.x; a.h[5]  = (__bf16)f1.y;
    a.h[6]  = (__bf16)f1.z; a.h[7]  = (__bf16)f1.w;
    a.h[8]  = (__bf16)f2.x; a.h[9]  = (__bf16)f2.y;
    a.h[10] = (__bf16)f2.z; a.h[11] = (__bf16)f2.w;
    a.h[12] = (__bf16)f3.x; a.h[13] = (__bf16)f3.y;
    a.h[14] = (__bf16)f3.z; a.h[15] = (__bf16)f3.w;

    const __bf16* wl = &Wl[buf][0];
    // B fragment (32x16) for i = m*4+t: N = h = i*16+ln, K = 16*hf + 2j,2j+1
    auto bcol = [&](int i) { return wl + ((i * 16 + ln) * 32 + hf * 16); };

    BF16Frag bfr[2];
    #pragma unroll
    for (int j = 0; j < 8; ++j)
      bfr[0].u[j] = *(const uint32_t*)(bcol(0) + 2 * j);
    #pragma unroll
    for (int i = 0; i < 12; ++i) {
      if (i + 1 < 12) {                  // prefetch next B fragment
        const __bf16* c = bcol(i + 1);
        #pragma unroll
        for (int j = 0; j < 8; ++j)
          bfr[(i + 1) & 1].u[j] = *(const uint32_t*)(c + 2 * j);
      }
      acc[i] = __builtin_amdgcn_wmma_f32_16x16x32_bf16(
          false, a.v, false, bfr[i & 1].v, (short)0, acc[i], false, false);
    }
    wait_async0();     // prefetch landed
    __syncthreads();   // all waves done reading Wl[buf]
    buf ^= 1;
  }

  const int b  = row0 / kT;
  const int tl = row0 % kT;
  #pragma unroll
  for (int t = 0; t < 4; ++t) {
    #pragma unroll
    for (int j = 0; j < 8; ++j) {
      // C/D layout: row = j + 8*hf, col = t*16 + ln
      int rowl = j + hf * 8;
      int grow = row0 + rowl;
      int h    = t * 16 + ln;
      Q[(size_t)grow * kH + h] = (__bf16)acc[0 * 4 + t][j];
      K[(size_t)grow * kH + h] = (__bf16)acc[1 * 4 + t][j];
      Vt[((size_t)b * kH + h) * kT + (tl + rowl)] = (__bf16)acc[2 * 4 + t][j];
    }
  }
}

// ---------------------------------------------------------------------------
// Kernel 2: causal flash attention.  4 waves/block = 64 consecutive queries;
// K/V tiles for each 32-key block async-staged in LDS (double-buffered) and
// shared by the 4 waves.  Block-uniform trip count; waves past their causal
// diagonal skip compute but keep staging and barriers.
// ---------------------------------------------------------------------------
__global__ __launch_bounds__(128) void attn_kernel(
    const __bf16* __restrict__ Q, const __bf16* __restrict__ K,
    const __bf16* __restrict__ Vt, float* __restrict__ out) {
  __shared__ __bf16 Kl[2][32 * 64];     // keys-major [key][h], 2 x 4 KB
  __shared__ __bf16 Vl[2][64 * 32];     // h-major [h][key],   2 x 4 KB
  __shared__ __bf16 psh[4][16 * 32];    // per-wave P transpose staging

  const int tid  = threadIdx.x;
  const int lane = tid & 31;
  const int wave = tid >> 5;
  const int hf   = lane >> 4;
  const int ln   = lane & 15;
  const int b    = (blockIdx.x * 64) / kT;
  const int qblk = (blockIdx.x * 64) % kT;    // block's first query
  const int q0   = qblk + wave * 16;          // this wave's first query

  // stage K[kb0..kb0+31][0..63] and Vt[0..63][kb0..kb0+31] -> LDS
  auto stage_kv = [&](int buf, int kb0) {
    const __bf16* ksrc = K + (size_t)(b * kT + kb0) * kH;   // 4 KB contiguous
    #pragma unroll
    for (int r = 0; r < 2; ++r) {
      int idx = tid + 128 * r;          // 0..255
      async_cp16(&Kl[buf][idx * 8], ksrc + idx * 8);
    }
    #pragma unroll
    for (int r = 0; r < 2; ++r) {
      int idx = tid + 128 * r;          // 0..255
      int h = idx >> 2, piece = idx & 3;
      async_cp16(&Vl[buf][h * 32 + piece * 8],
                 Vt + ((size_t)b * kH + h) * kT + kb0 + piece * 8);
    }
  };

  // Q fragments (A layout), K-dim = H = 64 -> two 16x32 fragments
  BF16Frag qa[2];
  {
    const __bf16* qrow = Q + (size_t)(b * kT + q0 + ln) * kH;
    #pragma unroll
    for (int f = 0; f < 2; ++f)
      #pragma unroll
      for (int j = 0; j < 8; ++j)
        qa[f].u[j] = *(const uint32_t*)(qrow + 32 * f + a_kbase(j, hf));
  }

  v8f o[4];
  #pragma unroll
  for (int t = 0; t < 4; ++t) o[t] = v8f{0.f, 0.f, 0.f, 0.f, 0.f, 0.f, 0.f, 0.f};
  float mrow[8], lrow[8];
  #pragma unroll
  for (int j = 0; j < 8; ++j) { mrow[j] = -__builtin_inff(); lrow[j] = 0.f; }
  const float scale = 0.125f;  // H^-0.5

  stage_kv(0, 0);
  wait_async0();
  __syncthreads();

  int buf = 0;
  for (int kb0 = 0; kb0 < qblk + 64; kb0 += 32) {
    if (kb0 + 32 < qblk + 64) stage_kv(buf ^ 1, kb0 + 32);  // prefetch

    if (kb0 <= q0 + 15) {   // wave-uniform: skip blocks fully above diagonal
      const __bf16* kl = &Kl[buf][0];
      const __bf16* vl = &Vl[buf][0];

      // ---- S = (Q K^T): load all 4 K fragments up-front, then 4 WMMAs
      BF16Frag kf[2][2];
      #pragma unroll
      for (int s = 0; s < 2; ++s) {
        const __bf16* krow = kl + ((s * 16 + ln) * 64 + hf * 16);
        #pragma unroll
        for (int f = 0; f < 2; ++f)
          #pragma unroll
          for (int j = 0; j < 8; ++j)
            kf[s][f].u[j] = *(const uint32_t*)(krow + 32 * f + 2 * j);
      }
      v8f S[2];
      #pragma unroll
      for (int s = 0; s < 2; ++s) {
        v8f st = v8f{0.f, 0.f, 0.f, 0.f, 0.f, 0.f, 0.f, 0.f};
        st = __builtin_amdgcn_wmma_f32_16x16x32_bf16(false, qa[0].v, false, kf[s][0].v,
                                                     (short)0, st, false, false);
        st = __builtin_amdgcn_wmma_f32_16x16x32_bf16(false, qa[1].v, false, kf[s][1].v,
                                                     (short)0, st, false, false);
        S[s] = st;
      }
      // ---- scale + causal mask (C layout: row = j+8*hf, col = ln)
      #pragma unroll
      for (int s = 0; s < 2; ++s)
        #pragma unroll
        for (int j = 0; j < 8; ++j) {
          int qg = q0 + j + hf * 8;
          int kg = kb0 + s * 16 + ln;
          float v = S[s][j] * scale;
          S[s][j] = (kg <= qg) ? v : -__builtin_inff();
        }
      // ---- online softmax across the 16-lane half-wave
      float newm[8], alpha[8];
      #pragma unroll
      for (int j = 0; j < 8; ++j) {
        float v = fmaxf(S[0][j], S[1][j]);
        #pragma unroll
        for (int d = 1; d < 16; d <<= 1) v = fmaxf(v, __shfl_xor(v, d, 32));
        newm[j]  = fmaxf(mrow[j], v);
        alpha[j] = __expf(mrow[j] - newm[j]);
        mrow[j]  = newm[j];
      }
      #pragma unroll
      for (int j = 0; j < 8; ++j) {
        float p0 = __expf(S[0][j] - newm[j]);
        float p1 = __expf(S[1][j] - newm[j]);
        S[0][j] = p0; S[1][j] = p1;
        float rs = p0 + p1;
        #pragma unroll
        for (int d = 1; d < 16; d <<= 1) rs += __shfl_xor(rs, d, 32);
        lrow[j] = lrow[j] * alpha[j] + rs;
      }
      #pragma unroll
      for (int t = 0; t < 4; ++t)
        #pragma unroll
        for (int j = 0; j < 8; ++j) o[t][j] *= alpha[j];

      // ---- P: C layout f32 -> LDS -> A layout bf16 (wave-local, no barrier)
      __bf16* ps = &psh[wave][0];
      #pragma unroll
      for (int j = 0; j < 8; ++j) {
        int row = j + hf * 8;
        ps[row * 32 + ln]      = (__bf16)S[0][j];
        ps[row * 32 + 16 + ln] = (__bf16)S[1][j];
      }
      asm volatile("s_wait_dscnt 0x0" ::: "memory");  // CDNA5 split DS counter
      BF16Frag pa;
      const uint32_t* ps32 = (const uint32_t*)ps;
      #pragma unroll
      for (int j = 0; j < 8; ++j)
        pa.u[j] = ps32[(ln * 32 + a_kbase(j, hf)) >> 1];

      // ---- O += P . V  (V h-major in LDS; fragments double-buffered)
      auto vcol = [&](int t) { return vl + ((t * 16 + ln) * 32 + hf * 16); };
      BF16Frag vb[2];
      #pragma unroll
      for (int j = 0; j < 8; ++j)
        vb[0].u[j] = *(const uint32_t*)(vcol(0) + 2 * j);
      #pragma unroll
      for (int t = 0; t < 4; ++t) {
        if (t + 1 < 4) {
          const __bf16* c = vcol(t + 1);
          #pragma unroll
          for (int j = 0; j < 8; ++j)
            vb[(t + 1) & 1].u[j] = *(const uint32_t*)(c + 2 * j);
        }
        o[t] = __builtin_amdgcn_wmma_f32_16x16x32_bf16(
            false, pa.v, false, vb[t & 1].v, (short)0, o[t], false, false);
      }
    }

    wait_async0();     // my prefetch writes landed
    __syncthreads();   // all waves done with current buffers
    buf ^= 1;
  }

  // ---- normalize and store f32 output [B,T,H]
  float rl[8];
  #pragma unroll
  for (int j = 0; j < 8; ++j) rl[j] = 1.0f / lrow[j];
  float* orow = out + (size_t)(b * kT + q0) * kH;
  #pragma unroll
  for (int t = 0; t < 4; ++t)
    #pragma unroll
    for (int j = 0; j < 8; ++j)
      orow[(size_t)(j + hf * 8) * kH + t * 16 + ln] = o[t][j] * rl[j];
}

// ---------------------------------------------------------------------------
extern "C" void kernel_launch(void* const* d_in, const int* in_sizes, int n_in,
                              void* d_out, int out_size, void* d_ws, size_t ws_size,
                              hipStream_t stream) {
  (void)in_sizes; (void)n_in; (void)out_size; (void)ws_size;
  const float* x  = (const float*)d_in[0];
  const float* Wq = (const float*)d_in[1];
  const float* Wk = (const float*)d_in[2];
  const float* Wv = (const float*)d_in[3];
  float* out = (float*)d_out;

  char* ws = (char*)d_ws;
  __bf16* Wt = (__bf16*)ws;                                    // 3*H*D bf16
  __bf16* Q  = (__bf16*)(ws + (size_t)3 * kH * kD * 2);        // B*T*H bf16
  __bf16* K  = Q + (size_t)kB * kT * kH;
  __bf16* Vt = K + (size_t)kB * kT * kH;                       // [B,H,T]

  wt_kernel<<<(3 * kD * kH + 255) / 256, 256, 0, stream>>>(Wq, Wk, Wv, Wt);

  // B*T/16 = 512 token tiles, 8 waves (256 thr) per block
  qkv_kernel<<<(kB * kT / 16) / 8, 256, 0, stream>>>(x, Wt, Q, K, Vt);

  // 64 queries per block (4 waves), B*T/64 = 128 blocks
  attn_kernel<<<(kB * kT) / 64, 128, 0, stream>>>(Q, K, Vt, out);
}
